// MultiheadAttention_37993280700997
// MI455X (gfx1250) — matrix operand order repaired and checked by reading, not compiled, not run on hardware
//
#include <hip/hip_runtime.h>
#include <hip/hip_bf16.h>
#include <math.h>

#define TGT   256
#define BSZ   64
#define EDIM  1024
#define NH    16
#define HD    64
#define PFX   16
#define TAUF  3.0f
#define SCALE 0.125f   // 1/sqrt(64)

typedef __attribute__((ext_vector_type(16))) __bf16 v16bf;
typedef __attribute__((ext_vector_type(8)))  float  v8f;

union Frag16 {
    uint4 q[2];
    unsigned int ui[8];
    unsigned short us[16];
    v16bf b;
};

// f32 -> bf16 by truncation (1 VALU)
__device__ inline unsigned short f2bf(float f) {
    return (unsigned short)(__builtin_bit_cast(unsigned int, f) >> 16);
}

// packed f32x2 -> bf16x2
__device__ inline unsigned int f2bf2(float lo, float hi) {
#if __has_builtin(__builtin_amdgcn_cvt_pk_bf16_f32)
    typedef __attribute__((ext_vector_type(2))) __bf16 v2bf;
    v2bf r = __builtin_amdgcn_cvt_pk_bf16_f32(lo, hi);
    return __builtin_bit_cast(unsigned int, r);
#elif __has_builtin(__builtin_amdgcn_perm)
    // v_perm_b32: out[1:0]=lo bytes 3:2, out[3:2]=hi bytes 3:2  (1 VALU)
    return __builtin_amdgcn_perm(__builtin_bit_cast(unsigned int, hi),
                                 __builtin_bit_cast(unsigned int, lo),
                                 0x07060302u);
#else
    return ((__builtin_bit_cast(unsigned int, hi) & 0xFFFF0000u) |
            (__builtin_bit_cast(unsigned int, lo) >> 16));
#endif
}

__device__ inline v8f wmma_bf16(v16bf a, v16bf b, v8f c) {
    return __builtin_amdgcn_wmma_f32_16x16x32_bf16(
        false, a, false, b, (short)0, c, false, false);
}

// async global->LDS copy path (CDNA5 ASYNCcnt), if toolchain exposes it.
// Probe-learned signature: param0 = int __attribute__((vector_size(16)))
// __device__* (target AS1, non-const); param1 = LDS (AS3) counterpart.
#if __has_builtin(__builtin_amdgcn_global_load_async_to_lds_b128) && \
    __has_builtin(__builtin_amdgcn_s_wait_asynccnt)
#define USE_ASYNC_LDS 1
typedef int v4i_vs __attribute__((vector_size(16)));
typedef __attribute__((address_space(1))) v4i_vs* as1_v4i;
typedef __attribute__((address_space(3))) v4i_vs* as3_v4i;
#define AS1V(p) ((as1_v4i)(p))
#define AS3V(p) ((as3_v4i)(p))
#else
#define USE_ASYNC_LDS 0
#endif

// ---------------------------------------------------------------------------
// Gumbel gate: gate1[b] = softmax((feat@gw.T + gb + noise)/tau)[1]
// ---------------------------------------------------------------------------
__global__ void gate_kernel(const float* __restrict__ feat,   // [64,512]
                            const float* __restrict__ gw,     // [2,512]
                            const float* __restrict__ gb,     // [2]
                            const float* __restrict__ noise,  // [64,2]
                            float* __restrict__ gate1) {      // [64]
    int j = threadIdx.x;
    if (j >= BSZ) return;
    const float* f = feat + j * 512;
    float l0 = gb[0], l1 = gb[1];
    for (int k = 0; k < 512; ++k) {
        float x = f[k];
        l0 = fmaf(x, gw[k], l0);
        l1 = fmaf(x, gw[512 + k], l1);
    }
    float z0 = (l0 + noise[j * 2 + 0]) / TAUF;
    float z1 = (l1 + noise[j * 2 + 1]) / TAUF;
    float mx = fmaxf(z0, z1);
    float e0 = __expf(z0 - mx), e1 = __expf(z1 - mx);
    gate1[j] = e1 / (e0 + e1);
}

// ---------------------------------------------------------------------------
// Generic GEMM: out[m,n] = A[m,:] . W[n,:] + bias[n]
// M=16384, N=1024, K=1024.  128x128 block tile, 8 wave32 waves (4x2),
// each wave 32x64 via 2x4 f32 16x16 WMMA accumulators, Kc=32 bf16 steps,
// double-buffered LDS (one barrier per K-step).
// mode 0: Q -> bf16 [b,h,t,d] | mode 1: K -> bf16 [b,h,s,d]
// mode 2: V -> bf16 [b,h,d,s] | mode 3: f32 row-major [m,n]
// ---------------------------------------------------------------------------
__global__ __launch_bounds__(256)
void proj_gemm_kernel(const float* __restrict__ A,
                      const float* __restrict__ W,
                      const float* __restrict__ bias,
                      float* __restrict__ outF,
                      unsigned short* __restrict__ outB,
                      int mode) {
    __shared__ unsigned short As[2][128 * 48];
    __shared__ unsigned short Bs[2][128 * 48];

    const int tid  = threadIdx.x;
    const int lane = tid & 31;
    const int wave = tid >> 5;
    const int wm   = wave >> 1;           // 0..3
    const int wn   = wave & 1;            // 0..1
    const int lh   = lane >> 4;           // lane half
    const int r16  = lane & 15;
    const int m0   = blockIdx.y * 128;
    const int n0   = blockIdx.x * 128;

    v8f acc[2][4];
    const v8f vzero = {0.f, 0.f, 0.f, 0.f, 0.f, 0.f, 0.f, 0.f};
    for (int i = 0; i < 2; ++i)
        for (int j = 0; j < 4; ++j) acc[i][j] = vzero;

    const int ldrow = tid >> 1;           // 0..127
    const int ldkc  = (tid & 1) * 16;     // 0 / 16
    const float* baseA = A + (size_t)(m0 + ldrow) * EDIM + ldkc;
    const float* baseB = W + (size_t)(n0 + ldrow) * EDIM + ldkc;

    float4 ra[4], rb[4];
    auto load_tiles = [&](int kb) {
        const float* sa = baseA + kb;
        const float* sb = baseB + kb;
        ra[0] = *(const float4*)(sa + 0);
        ra[1] = *(const float4*)(sa + 4);
        ra[2] = *(const float4*)(sa + 8);
        ra[3] = *(const float4*)(sa + 12);
        rb[0] = *(const float4*)(sb + 0);
        rb[1] = *(const float4*)(sb + 4);
        rb[2] = *(const float4*)(sb + 8);
        rb[3] = *(const float4*)(sb + 12);
        __builtin_prefetch(sa + 64, 0, 1);     // two tiles ahead
        __builtin_prefetch(sb + 64, 0, 1);
    };
    auto store_tiles = [&](int buf) {
        Frag16 fa, fb;
        fa.ui[0] = f2bf2(ra[0].x, ra[0].y); fa.ui[1] = f2bf2(ra[0].z, ra[0].w);
        fa.ui[2] = f2bf2(ra[1].x, ra[1].y); fa.ui[3] = f2bf2(ra[1].z, ra[1].w);
        fa.ui[4] = f2bf2(ra[2].x, ra[2].y); fa.ui[5] = f2bf2(ra[2].z, ra[2].w);
        fa.ui[6] = f2bf2(ra[3].x, ra[3].y); fa.ui[7] = f2bf2(ra[3].z, ra[3].w);
        fb.ui[0] = f2bf2(rb[0].x, rb[0].y); fb.ui[1] = f2bf2(rb[0].z, rb[0].w);
        fb.ui[2] = f2bf2(rb[1].x, rb[1].y); fb.ui[3] = f2bf2(rb[1].z, rb[1].w);
        fb.ui[4] = f2bf2(rb[2].x, rb[2].y); fb.ui[5] = f2bf2(rb[2].z, rb[2].w);
        fb.ui[6] = f2bf2(rb[3].x, rb[3].y); fb.ui[7] = f2bf2(rb[3].z, rb[3].w);
        *(uint4*)&As[buf][ldrow * 48 + ldkc]     = fa.q[0];
        *(uint4*)&As[buf][ldrow * 48 + ldkc + 8] = fa.q[1];
        *(uint4*)&Bs[buf][ldrow * 48 + ldkc]     = fb.q[0];
        *(uint4*)&Bs[buf][ldrow * 48 + ldkc + 8] = fb.q[1];
    };

    load_tiles(0);
    store_tiles(0);

    for (int kk = 0; kk < 32; ++kk) {
        __syncthreads();                       // buf[kk&1] ready, buf[kk^1] free
        if (kk + 1 < 32) load_tiles((kk + 1) * 32);   // issue early, hide under WMMA

        const unsigned short* Ab = As[kk & 1];
        const unsigned short* Bb = Bs[kk & 1];
        Frag16 afr[2], bfr[4];
        for (int i = 0; i < 2; ++i) {
            const unsigned short* p = &Ab[(wm * 32 + i * 16 + r16) * 48];
            afr[i].q[0] = *(const uint4*)(p + lh * 8);        // e0..7 : k = lh*8+e
            afr[i].q[1] = *(const uint4*)(p + 16 + lh * 8);   // e8..15: k = 16+lh*8+(e-8)
        }
        for (int j = 0; j < 4; ++j) {
            const unsigned short* p = &Bb[(wn * 64 + j * 16 + r16) * 48];
            bfr[j].q[0] = *(const uint4*)(p + lh * 8);        // e0..7 : k = lh*8+e
            bfr[j].q[1] = *(const uint4*)(p + 16 + lh * 8);   // e8..15: k = 16+lh*8+(e-8)
        }
        for (int i = 0; i < 2; ++i)
            for (int j = 0; j < 4; ++j)
                acc[i][j] = wmma_bf16(afr[i].b, bfr[j].b, acc[i][j]);

        if (kk + 1 < 32) store_tiles((kk + 1) & 1);
    }

    // ---- epilogue: bias + scatter per destination layout ----
    for (int i = 0; i < 2; ++i) {
        const int mbase = m0 + wm * 32 + i * 16 + lh * 8;
        for (int j = 0; j < 4; ++j) {
            const int n  = n0 + wn * 64 + j * 16 + r16;
            const float bv = bias[n];
            #pragma unroll
            for (int r = 0; r < 8; ++r) {
                const int m = mbase + r;
                const float v = acc[i][j][r] + bv;
                if (mode == 3) {
                    outF[(size_t)m * EDIM + n] = v;
                } else {
                    const int t = m >> 6, bb = m & 63;   // m = t*BSZ + b
                    const int h = n >> 6, d  = n & 63;   // n = h*HD + d
                    size_t idx;
                    if (mode == 2) idx = (((size_t)bb * NH + h) * HD + d) * TGT + t;
                    else           idx = (((size_t)bb * NH + h) * TGT + t) * HD + d;
                    outB[idx] = f2bf(v);
                }
            }
        }
    }
}

// ---------------------------------------------------------------------------
// Attention: block = (t-chunk of 32, head, batch). LDS holds K(+prefix) as
// bf16 [272][80], V^T(+prefix, zero-padded) as bf16 [64][288], scores f32
// [32][288]. Two independent softmaxes; prefix probs scaled by
// batch_weight[b]*gate1[t>>2]; writes combined f32 [t*BSZ+b][E].
// ---------------------------------------------------------------------------
#define KS_LD 80
#define VS_LD 288
#define SS_LD 288

__global__ __launch_bounds__(256)
void attn_kernel(const unsigned short* __restrict__ Qbf,   // [b,h,t,d]
                 const unsigned short* __restrict__ Kbf,   // [b,h,s,d]
                 const unsigned short* __restrict__ VTbf,  // [b,h,d,s]
                 const float* __restrict__ prefix,         // [b,2,16,1024]
                 const float* __restrict__ bweight,        // [64]
                 const float* __restrict__ gate1,          // [64]
                 float* __restrict__ comb) {               // [16384,1024]
    extern __shared__ char smem[];
    unsigned short* Ks  = (unsigned short*)smem;                              // 272*80
    unsigned short* VTs = (unsigned short*)(smem + 272 * KS_LD * 2);          // 64*288
    float*          S   = (float*)(smem + 272 * KS_LD * 2 + 64 * VS_LD * 2);  // 32*288

    const int tid  = threadIdx.x;
    const int lane = tid & 31;
    const int wave = tid >> 5;
    const int lh   = lane >> 4;
    const int r16  = lane & 15;
    const int t0   = blockIdx.x * 32;
    const int h    = blockIdx.y;
    const int bz   = blockIdx.z;
    const size_t bh = (size_t)bz * NH + h;

    // ---- stage K rows 0..255 and V^T rows (64 x 256) ----
#if USE_ASYNC_LDS
    {
        char* gk = (char*)(Kbf + (bh * TGT + tid) * HD);
        char* lk = (char*)&Ks[tid * KS_LD];
        #pragma unroll
        for (int i = 0; i < 8; ++i)
            __builtin_amdgcn_global_load_async_to_lds_b128(
                AS1V(gk + 16 * i), AS3V(lk + 16 * i), 0, 0);
        const int d = tid >> 2, sc = (tid & 3) * 64;
        char* gv = (char*)(VTbf + (bh * HD + d) * TGT + sc);
        char* lv = (char*)&VTs[d * VS_LD + sc];
        #pragma unroll
        for (int i = 0; i < 8; ++i)
            __builtin_amdgcn_global_load_async_to_lds_b128(
                AS1V(gv + 16 * i), AS3V(lv + 16 * i), 0, 0);
    }
#else
    {
        const uint4* src = (const uint4*)(Kbf + (bh * TGT + tid) * HD);
        uint4* dst = (uint4*)&Ks[tid * KS_LD];
        #pragma unroll
        for (int i = 0; i < 8; ++i) dst[i] = src[i];
    }
    {
        const int d = tid >> 2, sc = (tid & 3) * 64;
        const uint4* src = (const uint4*)(VTbf + (bh * HD + d) * TGT + sc);
        uint4* dst = (uint4*)&VTs[d * VS_LD + sc];
        #pragma unroll
        for (int i = 0; i < 8; ++i) dst[i] = src[i];
    }
#endif
    // ---- prefix pk/pv (raw, no projection) + zero pad cols 272..287 ----
    {
        const int p = tid >> 4, d4 = (tid & 15) * 4;
        const float* pk = prefix + (((size_t)bz * 2 + 0) * PFX + p) * EDIM + h * HD + d4;
        const float* pv = prefix + (((size_t)bz * 2 + 1) * PFX + p) * EDIM + h * HD + d4;
        float4 fk = *(const float4*)pk;
        float4 fv = *(const float4*)pv;
        unsigned int* kd = (unsigned int*)&Ks[(256 + p) * KS_LD + d4];
        kd[0] = f2bf2(fk.x, fk.y);
        kd[1] = f2bf2(fk.z, fk.w);
        VTs[(d4 + 0) * VS_LD + 256 + p] = f2bf(fv.x);
        VTs[(d4 + 1) * VS_LD + 256 + p] = f2bf(fv.y);
        VTs[(d4 + 2) * VS_LD + 256 + p] = f2bf(fv.z);
        VTs[(d4 + 3) * VS_LD + 256 + p] = f2bf(fv.w);
        const int dz = tid >> 2, pc = (tid & 3) * 4;
        #pragma unroll
        for (int i = 0; i < 4; ++i) VTs[dz * VS_LD + 272 + pc + i] = 0;
    }
#if USE_ASYNC_LDS
    __builtin_amdgcn_s_wait_asynccnt(0);
#endif
    __syncthreads();

    // ---- S = (Q K^T) * SCALE  via WMMA (272 cols incl. prefix) ----
    Frag16 qf[2][2];
    for (int mt = 0; mt < 2; ++mt) {
        const unsigned short* qrow = Qbf + (bh * TGT + (t0 + mt * 16 + r16)) * HD;
        #pragma unroll
        for (int ks = 0; ks < 2; ++ks) {
            qf[mt][ks].q[0] = *(const uint4*)(qrow + ks * 32 + lh * 8);
            qf[mt][ks].q[1] = *(const uint4*)(qrow + ks * 32 + 16 + lh * 8);
        }
    }
    for (int j = wave; j < 17; j += 8) {
        Frag16 kf[2];
        #pragma unroll
        for (int ks = 0; ks < 2; ++ks) {
            const unsigned short* p = &Ks[(j * 16 + r16) * KS_LD + ks * 32 + lh * 8];
            kf[ks].q[0] = *(const uint4*)p;
            kf[ks].q[1] = *(const uint4*)(p + 16);
        }
        for (int mt = 0; mt < 2; ++mt) {
            v8f c = {0.f, 0.f, 0.f, 0.f, 0.f, 0.f, 0.f, 0.f};
            c = wmma_bf16(qf[mt][0].b, kf[0].b, c);
            c = wmma_bf16(qf[mt][1].b, kf[1].b, c);
            const int rbase = mt * 16 + lh * 8;
            const int col   = j * 16 + r16;
            #pragma unroll
            for (int r = 0; r < 8; ++r) S[(rbase + r) * SS_LD + col] = c[r] * SCALE;
        }
    }
    __syncthreads();

    // ---- softmax: cols 0..255 and cols 256..271 independently ----
    {
        const int row = tid >> 3, g = tid & 7;
        float* Sr = &S[row * SS_LD];
        float mx = -1e30f;
        for (int c = g; c < 256; c += 8) mx = fmaxf(mx, Sr[c]);
        mx = fmaxf(mx, __shfl_xor(mx, 1));
        mx = fmaxf(mx, __shfl_xor(mx, 2));
        mx = fmaxf(mx, __shfl_xor(mx, 4));
        float sum = 0.f;
        for (int c = g; c < 256; c += 8) { float e = __expf(Sr[c] - mx); Sr[c] = e; sum += e; }
        sum += __shfl_xor(sum, 1); sum += __shfl_xor(sum, 2); sum += __shfl_xor(sum, 4);
        const float inv = 1.f / sum;
        for (int c = g; c < 256; c += 8) Sr[c] *= inv;

        float mx2 = -1e30f;
        for (int c = 256 + g; c < 272; c += 8) mx2 = fmaxf(mx2, Sr[c]);
        mx2 = fmaxf(mx2, __shfl_xor(mx2, 1));
        mx2 = fmaxf(mx2, __shfl_xor(mx2, 2));
        mx2 = fmaxf(mx2, __shfl_xor(mx2, 4));
        float sum2 = 0.f;
        for (int c = 256 + g; c < 272; c += 8) { float e = __expf(Sr[c] - mx2); Sr[c] = e; sum2 += e; }
        sum2 += __shfl_xor(sum2, 1); sum2 += __shfl_xor(sum2, 2); sum2 += __shfl_xor(sum2, 4);
        const int t = t0 + row;
        // combined = attn + gate1 * bw * attn_pre  (gate row index replicates
        // the reference's repeat-vs-(t,b) order mismatch -> gate1[t>>2])
        const float w = bweight[bz] * gate1[t >> 2] / sum2;
        for (int c = 256 + g; c < 272; c += 8) Sr[c] *= w;
        Sr[272 + g] = 0.f;        // pad so the last WMMA k-step is harmless
        Sr[280 + g] = 0.f;
    }
    __syncthreads();

    // ---- O = P @ V  (272 k-dim in 9 steps of 32) ----
    {
        const int mt = wave & 1;
        const int dt = wave >> 1;               // 0..3
        v8f o = {0.f, 0.f, 0.f, 0.f, 0.f, 0.f, 0.f, 0.f};
        for (int ks = 0; ks < 9; ++ks) {
            Frag16 af, bf;
            const float* sr = &S[(mt * 16 + r16) * SS_LD + ks * 32];
            float4 c0 = *(const float4*)(sr + lh * 8);
            float4 c1 = *(const float4*)(sr + lh * 8 + 4);
            float4 c2 = *(const float4*)(sr + 16 + lh * 8);
            float4 c3 = *(const float4*)(sr + 16 + lh * 8 + 4);
            af.ui[0] = f2bf2(c0.x, c0.y); af.ui[1] = f2bf2(c0.z, c0.w);
            af.ui[2] = f2bf2(c1.x, c1.y); af.ui[3] = f2bf2(c1.z, c1.w);
            af.ui[4] = f2bf2(c2.x, c2.y); af.ui[5] = f2bf2(c2.z, c2.w);
            af.ui[6] = f2bf2(c3.x, c3.y); af.ui[7] = f2bf2(c3.z, c3.w);
            const unsigned short* p = &VTs[(dt * 16 + r16) * VS_LD + ks * 32 + lh * 8];
            bf.q[0] = *(const uint4*)p;
            bf.q[1] = *(const uint4*)(p + 16);
            o = wmma_bf16(af.b, bf.b, o);
        }
        const int trow = t0 + mt * 16 + lh * 8;
        const int dcol = h * HD + dt * 16 + r16;
        #pragma unroll
        for (int r = 0; r < 8; ++r)
            comb[((size_t)(trow + r) * BSZ + bz) * EDIM + dcol] = o[r];
    }
}

// ---------------------------------------------------------------------------
extern "C" void kernel_launch(void* const* d_in, const int* in_sizes, int n_in,
                              void* d_out, int out_size, void* d_ws, size_t ws_size,
                              hipStream_t stream) {
    const float* query  = (const float*)d_in[1];
    const float* key    = (const float*)d_in[2];
    const float* value  = (const float*)d_in[3];
    const float* prefix = (const float*)d_in[4];
    const float* bw     = (const float*)d_in[5];
    const float* feat   = (const float*)d_in[6];
    const float* noise  = (const float*)d_in[7];
    const float* in_w   = (const float*)d_in[8];
    const float* in_b   = (const float*)d_in[9];
    const float* out_w  = (const float*)d_in[10];
    const float* out_b  = (const float*)d_in[11];
    const float* gw     = (const float*)d_in[12];
    const float* gb     = (const float*)d_in[13];
    float* out = (float*)d_out;

    // workspace carve-up
    const size_t NEL = (size_t)BSZ * NH * TGT * HD;      // 16,777,216
    char* ws = (char*)d_ws;
    unsigned short* Qbf  = (unsigned short*)ws;
    unsigned short* Kbf  = Qbf + NEL;
    unsigned short* VTbf = Kbf + NEL;
    float* comb  = (float*)(ws + 3 * NEL * sizeof(unsigned short));
    float* gate1 = (float*)((char*)comb + (size_t)TGT * BSZ * EDIM * sizeof(float));

    gate_kernel<<<1, 64, 0, stream>>>(feat, gw, gb, noise, gate1);

    dim3 gg(EDIM / 128, (TGT * BSZ) / 128);   // (8, 128)
    dim3 blk(256);
    proj_gemm_kernel<<<gg, blk, 0, stream>>>(query, in_w,               in_b,        nullptr, Qbf, 0);
    proj_gemm_kernel<<<gg, blk, 0, stream>>>(key,   in_w + 1024 * 1024, in_b + 1024, nullptr, Kbf, 1);
    proj_gemm_kernel<<<gg, blk, 0, stream>>>(value, in_w + 2*1024*1024, in_b + 2048, nullptr, VTbf, 2);

    size_t lds = 272 * KS_LD * 2 + 64 * VS_LD * 2 + 32 * SS_LD * 4;   // 117,248 B
    attn_kernel<<<dim3(TGT / 32, NH, BSZ), blk, lds, stream>>>(
        Qbf, Kbf, VTbf, prefix, bw, gate1, comb);

    proj_gemm_kernel<<<gg, blk, 0, stream>>>(comb, out_w, out_b, out, nullptr, 3);
}
